// Seq2SeqGRU_74921409511830
// MI455X (gfx1250) — matrix operand order repaired and checked by reading, not compiled, not run on hardware
//
#include <hip/hip_runtime.h>

// ---------------------------------------------------------------------------
// Seq2Seq GRU for MI455X (gfx1250, wave32, WMMA).
//
// Strategy: batch-partition the recurrence (rows independent) into 16
// persistent workgroups of 16 batch rows. Hidden states live in LDS as bf16
// (WMMA A operand); weights are pre-packed into WMMA B-fragment order (bf16)
// in d_ws and streamed from the 192MB L2 every timestep. Gates are computed
// with v_wmma_f32_16x16x32_bf16; r/z accumulate gi+gh in one accumulator,
// n keeps gi/gh split (needed for n = tanh(i_n + r*h_n)). Nonlinearity is
// fully register-resident per lane using the 16x16 f32 C/D layout.
// ---------------------------------------------------------------------------

#define BATCH 256
#define TLEN  512
#define HID   512
#define G3    1536      // 3*H gate width
#define OUTL  10
#define BT    16        // batch rows per workgroup
#define NWG   (BATCH / BT)
#define LDH   520       // padded LDS row stride (bf16 elems): 1040B rotates banks
#define ELEMS_PER_TILE 512                 // 16(N) x 32(K) bf16 per tile
#define TILES_PER_MAT  (96 * 16)           // 96 N-tiles x 16 K-tiles
#define MAT_ELEMS      (TILES_PER_MAT * ELEMS_PER_TILE)   // 786432 bf16

typedef __attribute__((ext_vector_type(16))) __bf16 v16bf;
typedef __attribute__((ext_vector_type(8)))  __bf16 v8bf;
typedef __attribute__((ext_vector_type(8)))  float  v8f;

// ----- bf16 <-> f32 via bit ops (avoid relying on __bf16 arithmetic) -------
__device__ __forceinline__ __bf16 f2bf(float f) {
    unsigned u = __float_as_uint(f);
    unsigned r = (u + 0x7FFFu + ((u >> 16) & 1u)) >> 16;   // RNE
    return __builtin_bit_cast(__bf16, (unsigned short)r);
}
__device__ __forceinline__ float bf2f(__bf16 b) {
    unsigned u = ((unsigned)__builtin_bit_cast(unsigned short, b)) << 16;
    return __uint_as_float(u);
}

__device__ __forceinline__ float sigm(float x) {
    return 1.0f / (1.0f + __expf(-x));
}
__device__ __forceinline__ float tanh_(float x) {
    x = fminf(fmaxf(x, -15.0f), 15.0f);
    float e = __expf(2.0f * x);
    return (e - 1.0f) / (e + 1.0f);
}

// ---------------------------------------------------------------------------
// Pack a (1536 x 512) f32 row-major weight W into WMMA B-fragment order:
// B[k][n] = W[n][k].  Tile (nt,kt): 512 bf16 ordered [lane][e] with
//   n = nt*16 + (lane&15),  k = kt*32 + e + 16*(lane>=16)
// so a wave's B fragment is one contiguous 32B load per lane.
// ---------------------------------------------------------------------------
__global__ void pack_weights(const float* __restrict__ w0, const float* __restrict__ w1,
                             const float* __restrict__ w2, const float* __restrict__ w3,
                             const float* __restrict__ w4, const float* __restrict__ w5,
                             unsigned short* __restrict__ dst_u16) {
    __bf16* dst = (__bf16*)dst_u16;
    const float* srcs[6] = {w0, w1, w2, w3, w4, w5};
    long total = 6L * MAT_ELEMS;
    for (long idx = (long)blockIdx.x * blockDim.x + threadIdx.x; idx < total;
         idx += (long)gridDim.x * blockDim.x) {
        int mat  = (int)(idx / MAT_ELEMS);
        int rem  = (int)(idx % MAT_ELEMS);
        int tile = rem >> 9;            // /512
        int lane = (rem >> 4) & 31;
        int e    = rem & 15;
        int nt   = tile >> 4;
        int kt   = tile & 15;
        int n    = nt * 16 + (lane & 15);
        int k    = kt * 32 + e + ((lane >> 4) << 4);
        dst[idx] = f2bf(srcs[mat][n * HID + k]);
    }
}

// ---------------------------------------------------------------------------
// One (16 x 1536) += (16 x 512) x (512 x 1536) gate GEMM contribution.
// Wave wv owns gate columns [64*wv, 64*wv+64) of each region r/z/n:
//   N-tile (region g, sub i): nt = g*32 + wv*4 + i.
// accN is the region-2 accumulator set (gi_n or gh_n depending on caller).
// ---------------------------------------------------------------------------
__device__ __forceinline__ void gru_gemm(const __bf16* __restrict__ sA,     // [16][LDH]
                                         const __bf16* __restrict__ wp,     // packed matrix
                                         int wv, int lane,
                                         v8f accR[4], v8f accZ[4], v8f accN[4]) {
    const int hi = lane >> 4;
    const int ln = lane & 15;
#pragma unroll 2
    for (int kt = 0; kt < 16; ++kt) {
        const int k0 = kt * 32;
        // A fragment (16x32 bf16): lane row m=ln; one contiguous 16-elem run at k0+16*hi.
        const __bf16* ap = sA + ln * LDH + k0 + hi * 16;
        v8bf a0 = *(const v8bf*)(ap);
        v8bf a1 = *(const v8bf*)(ap + 8);
        v16bf A = __builtin_shufflevector(a0, a1, 0, 1, 2, 3, 4, 5, 6, 7,
                                          8, 9, 10, 11, 12, 13, 14, 15);
#pragma unroll
        for (int g = 0; g < 3; ++g) {
#pragma unroll
            for (int i = 0; i < 4; ++i) {
                const int nt = g * 32 + wv * 4 + i;
                const v16bf B = *(const v16bf*)(wp + ((nt * 16 + kt) << 9) + lane * 16);
                v8f& acc = (g == 0) ? accR[i] : ((g == 1) ? accZ[i] : accN[i]);
                acc = __builtin_amdgcn_wmma_f32_16x16x32_bf16(
                    false, A, false, B, (short)0, acc, false, false);
            }
        }
    }
}

__device__ __forceinline__ void init_bias(const float* bi, const float* bh, int wv, int ln,
                                          v8f aR[4], v8f aZ[4], v8f aGN[4], v8f aHN[4]) {
#pragma unroll
    for (int i = 0; i < 4; ++i) {
        int c = wv * 64 + i * 16 + ln;
        float br  = bi[c] + bh[c];
        float bz  = bi[512 + c] + bh[512 + c];
        float bgn = bi[1024 + c];
        float bhn = bh[1024 + c];
#pragma unroll
        for (int e = 0; e < 8; ++e) {
            aR[i][e] = br; aZ[i][e] = bz; aGN[i][e] = bgn; aHN[i][e] = bhn;
        }
    }
}

// GRU nonlinearity + h update, fully in registers per lane; writes bf16 h'.
// f32 C/D layout: element (vgpr e, lane) -> m = e + 8*(lane>=16), c = lane&15.
__device__ __forceinline__ void gru_nonlin_store(__bf16* sH, int wv, int lane,
                                                 v8f aR[4], v8f aZ[4],
                                                 v8f aGN[4], v8f aHN[4]) {
    const int hi = lane >> 4;
    const int ln = lane & 15;
#pragma unroll
    for (int i = 0; i < 4; ++i) {
        int c = wv * 64 + i * 16 + ln;
#pragma unroll
        for (int e = 0; e < 8; ++e) {
            int m   = e + hi * 8;
            float r = sigm(aR[i][e]);
            float z = sigm(aZ[i][e]);
            float n = tanh_(aGN[i][e] + r * aHN[i][e]);
            float hp = bf2f(sH[m * LDH + c]);
            sH[m * LDH + c] = f2bf((1.0f - z) * n + z * hp);
        }
    }
}

// ---------------------------------------------------------------------------
// Persistent kernel: 16 WGs x 256 threads (8 wave32). One WG = 16 batch rows.
// ---------------------------------------------------------------------------
__global__ void __launch_bounds__(256)
seq2seq_gru(const float* __restrict__ x,
            const float* __restrict__ wih0, const float* __restrict__ bih0,
            const float* __restrict__ bhh0, const float* __restrict__ bih1,
            const float* __restrict__ bhh1,
            const float* __restrict__ gam, const float* __restrict__ bet,
            const float* __restrict__ dwih0, const float* __restrict__ dbih0,
            const float* __restrict__ dbhh0, const float* __restrict__ dbih1,
            const float* __restrict__ dbhh1,
            const float* __restrict__ fcw, const float* __restrict__ fcb,
            const unsigned short* __restrict__ wpack_u16,
            float* __restrict__ out) {
    __shared__ __bf16 sH0[BT][LDH];
    __shared__ __bf16 sH1[BT][LDH];
    __shared__ float  sBih0[G3], sBhh0[G3], sBih1[G3], sBhh1[G3];
    __shared__ float  sDih0[G3], sDhh0[G3], sDih1[G3], sDhh1[G3];
    __shared__ float  sW0[G3 * 2];   // enc w_ih0 (1536 x 2)
    __shared__ float  sWd0[G3];      // dec w_ih0 (1536 x 1)
    __shared__ float  sFw[HID], sGam[HID], sBet[HID];
    __shared__ float  sX[BT][2];
    __shared__ float  sY[BT];

    const int tid  = threadIdx.x;
    const int wv   = tid >> 5;
    const int lane = tid & 31;
    const int hi   = lane >> 4;
    const int ln   = lane & 15;
    const int b0   = blockIdx.x * BT;

    for (int i = tid; i < G3; i += 256) {
        sBih0[i] = bih0[i];  sBhh0[i] = bhh0[i];
        sBih1[i] = bih1[i];  sBhh1[i] = bhh1[i];
        sDih0[i] = dbih0[i]; sDhh0[i] = dbhh0[i];
        sDih1[i] = dbih1[i]; sDhh1[i] = dbhh1[i];
        sW0[2 * i] = wih0[2 * i]; sW0[2 * i + 1] = wih0[2 * i + 1];
        sWd0[i] = dwih0[i];
    }
    for (int i = tid; i < HID; i += 256) { sFw[i] = fcw[i]; sGam[i] = gam[i]; sBet[i] = bet[i]; }
    for (int i = tid; i < BT * LDH; i += 256) {
        (&sH0[0][0])[i] = f2bf(0.0f);
        (&sH1[0][0])[i] = f2bf(0.0f);
    }
    const float fcb0 = fcb[0];
    __syncthreads();

    const __bf16* wp     = (const __bf16*)wpack_u16;
    const __bf16* whh0p  = wp + 0L * MAT_ELEMS;
    const __bf16* wih1p  = wp + 1L * MAT_ELEMS;
    const __bf16* whh1p  = wp + 2L * MAT_ELEMS;
    const __bf16* dwhh0p = wp + 3L * MAT_ELEMS;
    const __bf16* dwih1p = wp + 4L * MAT_ELEMS;
    const __bf16* dwhh1p = wp + 5L * MAT_ELEMS;

    // ======================= Encoder: 512 timesteps =======================
    for (int t = 0; t < TLEN; ++t) {
        if (tid < 32) {      // stage x[b0..b0+15, t, 0:2] into LDS
            int m = tid >> 1, j = tid & 1;
            sX[m][j] = x[(long)(b0 + m) * (TLEN * 2) + t * 2 + j];
        }
        __syncthreads();

        v8f aR[4], aZ[4], aGN[4], aHN[4];

        // ---- layer 0: gi0 is a K=2 VALU projection folded into acc init ----
        {
            float x0r[8], x1r[8];
#pragma unroll
            for (int e = 0; e < 8; ++e) { int m = e + hi * 8; x0r[e] = sX[m][0]; x1r[e] = sX[m][1]; }
#pragma unroll
            for (int i = 0; i < 4; ++i) {
                int c = wv * 64 + i * 16 + ln;
                float wr0 = sW0[2 * c],          wr1 = sW0[2 * c + 1];
                float wz0 = sW0[2 * (512 + c)],  wz1 = sW0[2 * (512 + c) + 1];
                float wn0 = sW0[2 * (1024 + c)], wn1 = sW0[2 * (1024 + c) + 1];
                float br  = sBih0[c] + sBhh0[c];
                float bz  = sBih0[512 + c] + sBhh0[512 + c];
                float bgn = sBih0[1024 + c];
                float bhn = sBhh0[1024 + c];
#pragma unroll
                for (int e = 0; e < 8; ++e) {
                    aR[i][e]  = x0r[e] * wr0 + x1r[e] * wr1 + br;
                    aZ[i][e]  = x0r[e] * wz0 + x1r[e] * wz1 + bz;
                    aGN[i][e] = x0r[e] * wn0 + x1r[e] * wn1 + bgn;
                    aHN[i][e] = bhn;
                }
            }
        }
        gru_gemm(&sH0[0][0], whh0p, wv, lane, aR, aZ, aHN);     // gh0
        __syncthreads();                                        // all reads of sH0 done
        gru_nonlin_store(&sH0[0][0], wv, lane, aR, aZ, aGN, aHN);
        __syncthreads();

        // ---- layer 1: gi1 (A = h0_new) and gh1 (A = h1), r/z fused ----
        init_bias(sBih1, sBhh1, wv, ln, aR, aZ, aGN, aHN);
        gru_gemm(&sH0[0][0], wih1p, wv, lane, aR, aZ, aGN);     // gi1 (n -> aGN)
        gru_gemm(&sH1[0][0], whh1p, wv, lane, aR, aZ, aHN);     // gh1 (n -> aHN)
        __syncthreads();
        gru_nonlin_store(&sH1[0][0], wv, lane, aR, aZ, aGN, aHN);
        __syncthreads();
    }

    // ======================= LayerNorm of hT0, hT1 ========================
    // Wave wv handles rows 2wv, 2wv+1; lane covers 16 columns.
#pragma unroll
    for (int rr = 0; rr < 2; ++rr) {
        int m = wv * 2 + rr;
#pragma unroll
        for (int buf = 0; buf < 2; ++buf) {
            __bf16* hrow = (buf == 0) ? &sH0[m][0] : &sH1[m][0];
            float s = 0.0f, ss = 0.0f;
#pragma unroll
            for (int j = 0; j < 16; ++j) {
                float v = bf2f(hrow[lane * 16 + j]);
                s += v; ss += v * v;
            }
#pragma unroll
            for (int o = 16; o > 0; o >>= 1) {
                s  += __shfl_xor(s, o, 32);
                ss += __shfl_xor(ss, o, 32);
            }
            float mu  = s * (1.0f / 512.0f);
            float var = ss * (1.0f / 512.0f) - mu * mu;
            float inv = rsqrtf(var + 1e-5f);
#pragma unroll
            for (int j = 0; j < 16; ++j) {
                int cc = lane * 16 + j;
                float v = bf2f(hrow[cc]);
                hrow[cc] = f2bf((v - mu) * inv * sGam[cc] + sBet[cc]);
            }
        }
    }
    if (tid < BT) sY[tid] = x[(long)(b0 + tid) * (TLEN * 2) + (TLEN - 1) * 2 + 1];
    __syncthreads();

    // ===================== Decoder: 10 autoregressive steps ===============
    for (int st = 0; st < OUTL; ++st) {
        v8f aR[4], aZ[4], aGN[4], aHN[4];

        // ---- dec layer 0: scalar input y folded into acc init ----
        {
            float yr[8];
#pragma unroll
            for (int e = 0; e < 8; ++e) yr[e] = sY[e + hi * 8];
#pragma unroll
            for (int i = 0; i < 4; ++i) {
                int c = wv * 64 + i * 16 + ln;
                float wr = sWd0[c],        br  = sDih0[c] + sDhh0[c];
                float wz = sWd0[512 + c],  bz  = sDih0[512 + c] + sDhh0[512 + c];
                float wn = sWd0[1024 + c], bgn = sDih0[1024 + c], bhn = sDhh0[1024 + c];
#pragma unroll
                for (int e = 0; e < 8; ++e) {
                    aR[i][e]  = yr[e] * wr + br;
                    aZ[i][e]  = yr[e] * wz + bz;
                    aGN[i][e] = yr[e] * wn + bgn;
                    aHN[i][e] = bhn;
                }
            }
        }
        gru_gemm(&sH0[0][0], dwhh0p, wv, lane, aR, aZ, aHN);
        __syncthreads();
        gru_nonlin_store(&sH0[0][0], wv, lane, aR, aZ, aGN, aHN);
        __syncthreads();

        // ---- dec layer 1 ----
        init_bias(sDih1, sDhh1, wv, ln, aR, aZ, aGN, aHN);
        gru_gemm(&sH0[0][0], dwih1p, wv, lane, aR, aZ, aGN);
        gru_gemm(&sH1[0][0], dwhh1p, wv, lane, aR, aZ, aHN);
        __syncthreads();
        gru_nonlin_store(&sH1[0][0], wv, lane, aR, aZ, aGN, aHN);
        __syncthreads();

        // ---- FC head: y = h1 . fc_w + fc_b ; also next decoder input ----
#pragma unroll
        for (int rr = 0; rr < 2; ++rr) {
            int m = wv * 2 + rr;
            float s = 0.0f;
#pragma unroll
            for (int j = 0; j < 16; ++j) {
                int cc = lane * 16 + j;
                s += bf2f(sH1[m][cc]) * sFw[cc];
            }
#pragma unroll
            for (int o = 16; o > 0; o >>= 1) s += __shfl_xor(s, o, 32);
            if (lane == 0) {
                float yv = s + fcb0;
                sY[m] = yv;
                out[(long)(b0 + m) * OUTL + st] = yv;
            }
        }
        __syncthreads();
    }
}

// ---------------------------------------------------------------------------
extern "C" void kernel_launch(void* const* d_in, const int* in_sizes, int n_in,
                              void* d_out, int out_size, void* d_ws, size_t ws_size,
                              hipStream_t stream) {
    (void)in_sizes; (void)n_in; (void)out_size; (void)ws_size;

    const float* x     = (const float*)d_in[0];
    const float* wih0  = (const float*)d_in[1];
    const float* whh0  = (const float*)d_in[2];
    const float* bih0  = (const float*)d_in[3];
    const float* bhh0  = (const float*)d_in[4];
    const float* wih1  = (const float*)d_in[5];
    const float* whh1  = (const float*)d_in[6];
    const float* bih1  = (const float*)d_in[7];
    const float* bhh1  = (const float*)d_in[8];
    const float* gam   = (const float*)d_in[9];
    const float* bet   = (const float*)d_in[10];
    const float* dwih0 = (const float*)d_in[11];
    const float* dwhh0 = (const float*)d_in[12];
    const float* dbih0 = (const float*)d_in[13];
    const float* dbhh0 = (const float*)d_in[14];
    const float* dwih1 = (const float*)d_in[15];
    const float* dwhh1 = (const float*)d_in[16];
    const float* dbih1 = (const float*)d_in[17];
    const float* dbhh1 = (const float*)d_in[18];
    const float* fcw   = (const float*)d_in[19];
    const float* fcb   = (const float*)d_in[20];

    unsigned short* wpack = (unsigned short*)d_ws;   // 6 * 786432 bf16 = 9.4 MB

    long total = 6L * MAT_ELEMS;
    int  pblocks = (int)((total + 255) / 256);
    pack_weights<<<pblocks, 256, 0, stream>>>(whh0, wih1, whh1, dwhh0, dwih1, dwhh1, wpack);

    seq2seq_gru<<<NWG, 256, 0, stream>>>(x, wih0, bih0, bhh0, bih1, bhh1, gam, bet,
                                         dwih0, dbih0, dbhh0, dbih1, dbhh1, fcw, fcb,
                                         wpack, (float*)d_out);
}
